// GCN_23003844838068
// MI455X (gfx1250) — matrix operand order repaired and checked by reading, not compiled, not run on hardware
//
#include <hip/hip_runtime.h>

// ---------------------------------------------------------------------------
// GCN (2-layer, PyG GCNConv semantics) for MI455X / gfx1250.
//   h  = relu( Agg(x@W1) + (x@W1)/deg + b1 )
//   out =      Agg(h@W2) + (h@W2)/deg + b2
// GEMMs: v_wmma_f32_16x16x32_bf16, wave32, one 16x16 tile per wave.
// Aggregation: native global_atomic_add_f32 scatter (node features stay in
// the 192MB L2; 23.3TB/s HBM is never the limiter).
// ---------------------------------------------------------------------------

#define GCN_N   50000
#define GCN_E   800000
#define GCN_FIN 128
#define GCN_FH  128
#define GCN_FOUT 64

typedef __attribute__((ext_vector_type(16))) __bf16 v16bf;
typedef __attribute__((ext_vector_type(8)))  __bf16 v8bf;
typedef __attribute__((ext_vector_type(8)))  float  v8f;

__device__ __forceinline__ __bf16 f2bf(float f) {
  // round-to-nearest-even f32 -> bf16
  unsigned u = __builtin_bit_cast(unsigned, f);
  u += 0x7FFFu + ((u >> 16) & 1u);
  return __builtin_bit_cast(__bf16, (unsigned short)(u >> 16));
}

// --- init: deg = 1.0 (self loop), agg buffer = 0 ---------------------------
__global__ void gcn_init(float* __restrict__ deg, float* __restrict__ aggbuf,
                         int n, int nbuf) {
  int i = blockIdx.x * blockDim.x + threadIdx.x;
  if (i < nbuf) aggbuf[i] = 0.0f;
  if (i < n)    deg[i] = 1.0f;
}

__global__ void gcn_zero(float* __restrict__ p, int n) {
  int i = blockIdx.x * blockDim.x + threadIdx.x;
  if (i < n) p[i] = 0.0f;
}

// --- degree: deg[dst] += 1 --------------------------------------------------
__global__ void gcn_degree(const int* __restrict__ dst, float* __restrict__ deg,
                           int e) {
  int i = blockIdx.x * blockDim.x + threadIdx.x;
  if (i < e)
    __hip_atomic_fetch_add(&deg[dst[i]], 1.0f,
                           __ATOMIC_RELAXED, __HIP_MEMORY_SCOPE_AGENT);
}

// --- dinv = deg^-1/2, drec = deg^-1 ----------------------------------------
__global__ void gcn_invdeg(const float* __restrict__ deg,
                           float* __restrict__ dinv, float* __restrict__ drec,
                           int n) {
  int i = blockIdx.x * blockDim.x + threadIdx.x;
  if (i < n) {
    float d = deg[i];
    dinv[i] = rsqrtf(d);
    drec[i] = 1.0f / d;
  }
}

// --- f32 -> bf16 bulk convert ----------------------------------------------
__global__ void gcn_cvt_bf16(const float* __restrict__ in,
                             __bf16* __restrict__ out, int n) {
  int i = blockIdx.x * blockDim.x + threadIdx.x;
  if (i < n) out[i] = f2bf(in[i]);
}

// --- pack W (KxNOUT f32, row-major) into WMMA-B fragment order -------------
// Wp[((t*4 + s)*32 + lane)*16 + j] = bf16( W[(s*32 + (lane>=16)*8 + (j&7) + (j>=8)*16)*NOUT
//                                            + t*16 + (lane&15)] )
__global__ void gcn_pack_w(const float* __restrict__ W,
                           __bf16* __restrict__ Wp,
                           int nout, int total) {
  int g = blockIdx.x * blockDim.x + threadIdx.x;
  if (g >= total) return;
  int j    = g & 15;
  int lane = (g >> 4) & 31;
  int s    = (g >> 9) & 3;        // 4 k-steps (K = 128)
  int t    = g >> 11;             // n-tile
  int k = (s << 5) + ((lane >> 4) << 3) + (j & 7) + ((j >> 3) << 4);
  int n = (t << 4) + (lane & 15);
  Wp[g] = f2bf(W[k * nout + n]);
}

// --- GEMM: C[M x NOUT] = A[M x 128](bf16) * Wp(bf16 packed), f32 accum -----
// grid.x = M/16 slabs; blockDim.x = 32 * (NOUT/16); wave w owns n-tile w.
// A fragment (16-bit A 16x32, ISA 7.12.2): lane<16 -> row=lane, K chunks
// [k0..k0+7] & [k0+16..k0+23]; lanes 16-31 take the +8 chunks.
template <int NOUT>
__global__ __launch_bounds__(256) void gcn_gemm_wmma(
    const __bf16* __restrict__ A,   // M x 128 bf16 row-major
    const __bf16* __restrict__ Bp,  // packed B fragments
    float* __restrict__ C) {
  const int lane  = threadIdx.x & 31;
  const int ntile = threadIdx.x >> 5;
  const int mtile = blockIdx.x;

  const int rowA = (mtile << 4) + (lane & 15);
  const int sel8 = (lane >> 4) << 3;                 // 0 | 8
  const __bf16* arow  = A  + rowA * 128;
  const __bf16* bbase = Bp + (ntile << 11) + (lane << 4);

  v8f acc = {};
#pragma unroll
  for (int s = 0; s < 4; ++s) {
    v8bf alo = *(const v8bf*)(arow + (s << 5) + sel8);
    v8bf ahi = *(const v8bf*)(arow + (s << 5) + 16 + sel8);
    v8bf blo = *(const v8bf*)(bbase + (s << 9));
    v8bf bhi = *(const v8bf*)(bbase + (s << 9) + 8);
    v16bf a = __builtin_shufflevector(alo, ahi,
        0, 1, 2, 3, 4, 5, 6, 7, 8, 9, 10, 11, 12, 13, 14, 15);
    v16bf b = __builtin_shufflevector(blo, bhi,
        0, 1, 2, 3, 4, 5, 6, 7, 8, 9, 10, 11, 12, 13, 14, 15);
    acc = __builtin_amdgcn_wmma_f32_16x16x32_bf16(
        false, a, false, b, (short)0, acc, false, false);
  }

  // C/D layout: VGPR r -> row = 16*mtile + r + (lane>=16 ? 8 : 0), col = lane%16
  const int col   = (ntile << 4) + (lane & 15);
  const int rbase = (mtile << 4) + ((lane >> 4) << 3);
  float* crow = C + rbase * NOUT + col;
#pragma unroll
  for (int r = 0; r < 8; ++r)
    crow[r * NOUT] = acc[r];
}

// --- edge scatter: agg[dst,f] += dinv[src]*dinv[dst] * h[src,f] ------------
__global__ void gcn_agg(const int* __restrict__ src, const int* __restrict__ dst,
                        const float* __restrict__ dinv,
                        const float* __restrict__ h, float* __restrict__ agg,
                        int e, int fdim, int logf) {
  int tid = blockIdx.x * blockDim.x + threadIdx.x;
  if (tid >= (e << logf)) return;
  int ed = tid >> logf;
  int f  = tid & (fdim - 1);
  int s  = src[ed];
  int d  = dst[ed];
  float nrm = dinv[s] * dinv[d];
  float v   = h[s * fdim + f] * nrm;
  __hip_atomic_fetch_add(&agg[d * fdim + f], v,
                         __ATOMIC_RELAXED, __HIP_MEMORY_SCOPE_AGENT);
}

// --- combine layer 1: h = relu(agg + h0/deg + b1) -> bf16 ------------------
__global__ void gcn_combine1(const float* __restrict__ agg,
                             const float* __restrict__ h0,
                             const float* __restrict__ drec,
                             const float* __restrict__ bias,
                             __bf16* __restrict__ hb, int total) {
  int i = blockIdx.x * blockDim.x + threadIdx.x;
  if (i >= total) return;
  int n = i >> 7;            // FH = 128
  int f = i & 127;
  float v = agg[i] + h0[i] * drec[n] + bias[f];
  hb[i] = f2bf(fmaxf(v, 0.0f));
}

// --- combine layer 2: out = agg + g0/deg + b2 (f32) ------------------------
__global__ void gcn_combine2(const float* __restrict__ agg,
                             const float* __restrict__ g0,
                             const float* __restrict__ drec,
                             const float* __restrict__ bias,
                             float* __restrict__ out, int total) {
  int i = blockIdx.x * blockDim.x + threadIdx.x;
  if (i >= total) return;
  int n = i >> 6;            // FOUT = 64
  int f = i & 63;
  out[i] = agg[i] + g0[i] * drec[n] + bias[f];
}

// ---------------------------------------------------------------------------
static inline int cdiv_i(long long a, int b) { return (int)((a + b - 1) / b); }

extern "C" void kernel_launch(void* const* d_in, const int* in_sizes, int n_in,
                              void* d_out, int out_size, void* d_ws, size_t ws_size,
                              hipStream_t stream) {
  (void)in_sizes; (void)n_in; (void)out_size; (void)ws_size;

  const float* x  = (const float*)d_in[0];              // N x 128
  const int*   ei = (const int*)d_in[1];                // 2 x E
  const float* W1 = (const float*)d_in[2];              // 128 x 128
  const float* b1 = (const float*)d_in[3];              // 128
  const float* W2 = (const float*)d_in[4];              // 128 x 64
  const float* b2 = (const float*)d_in[5];              // 64
  float*       out = (float*)d_out;                     // N x 64

  const int* src = ei;
  const int* dst = ei + GCN_E;

  // ---- carve workspace ----
  size_t off = 0;
  auto carve = [&](size_t bytes) -> void* {
    off = (off + 255) & ~(size_t)255;
    void* p = (char*)d_ws + off;
    off += bytes;
    return p;
  };
  float*  deg  = (float*)carve((size_t)GCN_N * 4);
  float*  dinv = (float*)carve((size_t)GCN_N * 4);
  float*  drec = (float*)carve((size_t)GCN_N * 4);
  __bf16* xb   = (__bf16*)carve((size_t)GCN_N * GCN_FIN * 2);
  __bf16* hb   = (__bf16*)carve((size_t)GCN_N * GCN_FH * 2);
  __bf16* w1p  = (__bf16*)carve((size_t)8 * 4 * 32 * 16 * 2);
  __bf16* w2p  = (__bf16*)carve((size_t)4 * 4 * 32 * 16 * 2);
  float*  bufA = (float*)carve((size_t)GCN_N * GCN_FH * 4);  // h0 / g0
  float*  bufB = (float*)carve((size_t)GCN_N * GCN_FH * 4);  // agg1 / agg2

  const int BT = 256;
  const long long NH = (long long)GCN_N * GCN_FH;    // 6.4M
  const long long NO = (long long)GCN_N * GCN_FOUT;  // 3.2M
  const long long EH = (long long)GCN_E * GCN_FH;    // 102.4M
  const long long EO = (long long)GCN_E * GCN_FOUT;  // 51.2M

  // ---- graph prep ----
  gcn_init  <<<cdiv_i(NH, BT), BT, 0, stream>>>(deg, bufB, GCN_N, (int)NH);
  gcn_degree<<<cdiv_i(GCN_E, BT), BT, 0, stream>>>(dst, deg, GCN_E);
  gcn_invdeg<<<cdiv_i(GCN_N, BT), BT, 0, stream>>>(deg, dinv, drec, GCN_N);
  gcn_cvt_bf16<<<cdiv_i(NH, BT), BT, 0, stream>>>(x, xb, (int)NH);
  gcn_pack_w<<<cdiv_i(16384, BT), BT, 0, stream>>>(W1, w1p, GCN_FH, 16384);
  gcn_pack_w<<<cdiv_i(8192, BT), BT, 0, stream>>>(W2, w2p, GCN_FOUT, 8192);

  // ---- layer 1: h0 = x@W1 ; agg ; combine(+relu) -> hb(bf16) ----
  gcn_gemm_wmma<GCN_FH><<<GCN_N / 16, 32 * (GCN_FH / 16), 0, stream>>>(xb, w1p, bufA);
  gcn_agg<<<cdiv_i(EH, BT), BT, 0, stream>>>(src, dst, dinv, bufA, bufB,
                                             GCN_E, GCN_FH, 7);
  gcn_combine1<<<cdiv_i(NH, BT), BT, 0, stream>>>(bufB, bufA, drec, b1, hb, (int)NH);

  // ---- layer 2: g0 = h@W2 ; agg ; combine -> out ----
  gcn_zero<<<cdiv_i(NO, BT), BT, 0, stream>>>(bufB, (int)NO);
  gcn_gemm_wmma<GCN_FOUT><<<GCN_N / 16, 32 * (GCN_FOUT / 16), 0, stream>>>(hb, w2p, bufA);
  gcn_agg<<<cdiv_i(EO, BT), BT, 0, stream>>>(src, dst, dinv, bufA, bufB,
                                             GCN_E, GCN_FOUT, 6);
  gcn_combine2<<<cdiv_i(NO, BT), BT, 0, stream>>>(bufB, bufA, drec, b2, out, (int)NO);
}
